// CPC_85117661872355
// MI455X (gfx1250) — matrix operand order repaired and checked by reading, not compiled, not run on hardware
//
#include <hip/hip_runtime.h>
#include <stdint.h>

#define BB 64
#define GG 7
#define DD 1280
#define STEPS 5
#define NNEG 16
#define NPRED 8960   // sum over s of 448*(6-s)

typedef __attribute__((ext_vector_type(16))) __bf16 v16bf;
typedef __attribute__((ext_vector_type(8)))  float  v8f;
typedef __attribute__((ext_vector_type(4)))  unsigned int u32x4;
typedef __attribute__((ext_vector_type(8)))  int          i32x8;
typedef __attribute__((ext_vector_type(4)))  int          i32x4;

#if defined(__has_builtin)
#if __has_builtin(__builtin_amdgcn_tensor_load_to_lds) && __has_builtin(__builtin_amdgcn_s_wait_tensorcnt)
#define HAVE_TDM 1
#endif
#endif
#ifndef HAVE_TDM
#define HAVE_TDM 0
#endif

// ---------------------------------------------------------------------------
// round-to-nearest-even f32 -> bf16 (finite inputs)
__device__ __forceinline__ unsigned short f2bf(float f) {
    union { float f; unsigned int u; } x; x.f = f;
    unsigned int u = x.u;
    u += 0x7FFFu + ((u >> 16) & 1u);
    return (unsigned short)(u >> 16);
}

// packed prediction row m (0..8959) -> (s, b, r, c)
__device__ __forceinline__ void row_decode(int m, int& s, int& b, int& r, int& c) {
    int base;
    if      (m < 2688) { s = 0; base = 0; }
    else if (m < 4928) { s = 1; base = 2688; }
    else if (m < 6720) { s = 2; base = 4928; }
    else if (m < 8064) { s = 3; base = 6720; }
    else               { s = 4; base = 8064; }
    int p      = m - base;
    int per_b  = (6 - s) * 7;
    b = p / per_b;
    int rem = p - b * per_b;
    r = rem / 7;
    c = rem - r * 7;
}

// ---------------------------------------------------------------------------
// Kernel 1: convert Wk_w -> bf16 and build packed bf16 A matrix (8960 x 1280)
// Apack row m corresponds to contexts[b, r, c] per the step-s slicing.
__global__ __launch_bounds__(256)
void cvt_kernel(const float* __restrict__ W, const float* __restrict__ ctx,
                unsigned short* __restrict__ Wbf, unsigned short* __restrict__ Apk) {
    const size_t WEL = (size_t)STEPS * DD * DD;        // 8,192,000 (mult of 8)
    const size_t AEL = (size_t)NPRED * DD;             // 11,468,800 (mult of 8)
    size_t i8 = ((size_t)blockIdx.x * blockDim.x + threadIdx.x) * 8;
    const float* src; unsigned short* dst; size_t off;
    if (i8 < WEL) {
        src = W; dst = Wbf; off = i8;
    } else {
        size_t j = i8 - WEL;
        if (j >= AEL) return;
        int m   = (int)(j / DD);
        int col = (int)(j - (size_t)m * DD);
        int s, b, r, c;
        row_decode(m, s, b, r, c);
        src = ctx + (size_t)(b * 49 + r * 7 + c) * DD + col;
        dst = Apk + (size_t)m * DD + col;
        off = 0;
    }
    float4 a = *(const float4*)(src + off);
    float4 b4 = *(const float4*)(src + off + 4);
    union { unsigned short s[8]; uint4 q; } o;
    o.s[0] = f2bf(a.x);  o.s[1] = f2bf(a.y);  o.s[2] = f2bf(a.z);  o.s[3] = f2bf(a.w);
    o.s[4] = f2bf(b4.x); o.s[5] = f2bf(b4.y); o.s[6] = f2bf(b4.z); o.s[7] = f2bf(b4.w);
    *(uint4*)(dst + off) = o.q;
}

// ---------------------------------------------------------------------------
// TDM: 2D tile load global->LDS. Descriptor packing per CDNA5 ISA.
// data_size = 2 bytes; dims/strides in element units; lds_addr/global_addr bytes.
// Toolchain here declares the 6-arg builtin:
//   (uint32x4 g0, int32x8 g1, int32x4 g2, int32x4 g3, int32x8 extra, i32 cpol)
__device__ __forceinline__ void tdm_load_2d(unsigned lds_addr, unsigned long long gaddr,
                                            unsigned tensor_d0, unsigned tensor_d1,
                                            unsigned tile_d0, unsigned tile_d1,
                                            unsigned long long stride0) {
#if HAVE_TDM
    u32x4 g0;
    g0.x = 1u;                                              // count=1, user mode
    g0.y = lds_addr;                                        // LDS byte address
    g0.z = (unsigned)(gaddr & 0xFFFFFFFFull);               // global_addr[31:0]
    g0.w = (unsigned)((gaddr >> 32) & 0x01FFFFFFull)        // global_addr[56:32]
         | (2u << 30);                                      // type = 2 ("image")
    i32x8 g1;
    g1[0] = (int)(1u << 16);                                // data_size=1 (2B), mask=0
    g1[1] = (int)((tensor_d0 & 0xFFFFu) << 16);             // tensor_dim0[15:0]
    g1[2] = (int)((tensor_d0 >> 16) | ((tensor_d1 & 0xFFFFu) << 16));
    g1[3] = (int)((tensor_d1 >> 16) | ((tile_d0 & 0xFFFFu) << 16));
    g1[4] = (int)(tile_d1 & 0xFFFFu);                       // tile_dim2 = 0 (2D)
    g1[5] = (int)(unsigned)(stride0 & 0xFFFFFFFFull);       // tensor_dim0_stride
    g1[6] = (int)((unsigned)(stride0 >> 32) & 0xFFFFu);     // stride1 = 0
    g1[7] = 0;
    i32x4 z4 = {0, 0, 0, 0};
    i32x8 z8 = {0, 0, 0, 0, 0, 0, 0, 0};
    __builtin_amdgcn_tensor_load_to_lds(g0, g1, z4, z4, z8, 0);
#else
    (void)lds_addr; (void)gaddr; (void)tensor_d0; (void)tensor_d1;
    (void)tile_d0; (void)tile_d1; (void)stride0;
#endif
}

// ---------------------------------------------------------------------------
// Kernel 2: pred[s] = Apack_s @ W_s^T + b_s  via v_wmma_f32_16x16x32_bf16
// block = 256 threads (8 waves, 2x4); block tile 128x256; wave tile 64x64;
// K-step 32; double-buffered LDS fed by the Tensor Data Mover.
__global__ __launch_bounds__(256)
void gemm_kernel(const unsigned short* __restrict__ Apk,
                 const unsigned short* __restrict__ Wbf,
                 const float* __restrict__ Wb,
                 float* __restrict__ pred) {
    const int s      = blockIdx.z;
    const int rows_s = 6 - s;
    const int M_s    = 448 * rows_s;
    const int tileM  = blockIdx.y * 128;
    if (tileM >= M_s) return;
    const int tileN   = blockIdx.x * 256;
    const int mbase_s = 448 * (6 * s - (s * (s - 1)) / 2);  // cumulative rows
    float* __restrict__ predS = pred + (size_t)mbase_s * DD;

    __shared__ uint4 ldsA[2][128 * 4];   // 128 rows x 32 bf16 per buffer (8KB)
    __shared__ uint4 ldsB[2][256 * 4];   // 256 rows x 32 bf16 per buffer (16KB)

    const int tid  = threadIdx.x;
    const int lane = tid & 31;
    const int wid  = tid >> 5;
    const int wm   = wid >> 2;        // 0..1 : 64 rows
    const int wn   = wid & 3;         // 0..3 : 64 cols
    const int hf   = lane >> 4;
    const int l16  = lane & 15;

    const unsigned short* Asrc = Apk + (size_t)(mbase_s + tileM) * DD;
    const unsigned short* Bsrc = Wbf + ((size_t)s * DD + tileN) * DD;
    const unsigned remA = (unsigned)(M_s - tileM);          // TDM zero-fills rows >= remA

#if !HAVE_TDM
    const int ldRow = tid >> 1;
    const int ldCol = (tid & 1) * 2;
    const bool aValid = (tileM + ldRow) < M_s;
#endif

    v8f acc[4][4];
    #pragma unroll
    for (int i = 0; i < 4; i++)
        #pragma unroll
        for (int j = 0; j < 4; j++) { v8f z = {}; acc[i][j] = z; }

    // ---- stage(k0, buf): start filling LDS buffer `buf` with K-slice k0 ----
    auto stage = [&](int k0, int buf) {
#if HAVE_TDM
        if (wid == 0) {
            tdm_load_2d((unsigned)(size_t)&ldsA[buf][0],
                        (unsigned long long)(size_t)(Asrc + k0),
                        32u, remA, 32u, 128u, (unsigned long long)DD);
            tdm_load_2d((unsigned)(size_t)&ldsB[buf][0],
                        (unsigned long long)(size_t)(Bsrc + k0),
                        32u, 256u, 32u, 256u, (unsigned long long)DD);
        }
#else
        uint4 a0 = make_uint4(0, 0, 0, 0), a1 = a0;
        if (aValid) {
            const uint4* p = (const uint4*)(Asrc + (size_t)ldRow * DD + k0) + ldCol;
            a0 = p[0]; a1 = p[1];
        }
        ldsA[buf][ldRow * 4 + ldCol]     = a0;
        ldsA[buf][ldRow * 4 + ldCol + 1] = a1;
        const uint4* q = (const uint4*)(Bsrc + (size_t)tid * DD + k0);
        #pragma unroll
        for (int i = 0; i < 4; i++) ldsB[buf][tid * 4 + i] = q[i];
#endif
    };

    stage(0, 0);

    for (int it = 0; it < DD / 32; ++it) {
        const int buf = it & 1;
#if HAVE_TDM
        if (wid == 0) __builtin_amdgcn_s_wait_tensorcnt(0);  // buf[it&1] landed
#endif
        __syncthreads();                                     // publish; prev compute done
        if (it + 1 < DD / 32) stage((it + 1) * 32, buf ^ 1); // overlap next K-slice

        union Frag { v16bf v; uint4 q[2]; };
        const uint4* LA = ldsA[buf];
        const uint4* LB = ldsB[buf];
        // B frags: lanes 0-15 -> K 0..15 of col N = l16 ; lanes 16-31 -> K 16..31
        Frag fb[4];
        #pragma unroll
        for (int fn = 0; fn < 4; fn++) {
            int nl = wn * 64 + fn * 16 + l16;
            fb[fn].q[0] = LB[nl * 4 + 2 * hf];
            fb[fn].q[1] = LB[nl * 4 + 2 * hf + 1];
        }
        // A frags: half 0 -> K 0..15, half 1 -> K 16..31
        #pragma unroll
        for (int fm = 0; fm < 4; fm++) {
            Frag fa;
            int ml = wm * 64 + fm * 16 + l16;
            fa.q[0] = LA[ml * 4 + 2 * hf];
            fa.q[1] = LA[ml * 4 + 2 * hf + 1];
            #pragma unroll
            for (int fn = 0; fn < 4; fn++) {
                acc[fm][fn] = __builtin_amdgcn_wmma_f32_16x16x32_bf16(
                    false, fa.v, false, fb[fn].v, (short)0, acc[fm][fn],
                    false, false);
            }
        }
    }

    // fused bias add + store (C layout: VGPR v -> M = v + 8*half, N = l16)
    #pragma unroll
    for (int fn = 0; fn < 4; fn++) {
        int n = tileN + wn * 64 + fn * 16 + l16;
        float bv = Wb[s * DD + n];
        #pragma unroll
        for (int fm = 0; fm < 4; fm++) {
            int mb = tileM + wm * 64 + fm * 16 + hf * 8;
            #pragma unroll
            for (int v = 0; v < 8; v++) {
                int m = mb + v;
                if (m < M_s)
                    predS[(size_t)m * DD + n] = acc[fm][fn][v] + bv;
            }
        }
    }
}

// ---------------------------------------------------------------------------
// Kernel 3: per-row scoring. One wave per prediction row; 8 rows per block.
__global__ __launch_bounds__(256)
void score_kernel(const float* __restrict__ pred,
                  const float* __restrict__ enc,
                  const int* __restrict__ neg_idx,
                  float* __restrict__ partial) {
    const int tid  = threadIdx.x;
    const int lane = tid & 31;
    const int wid  = tid >> 5;
    const int m    = blockIdx.x * 8 + wid;   // 1120*8 == 8960 exactly

    int s, b, r, c;
    row_decode(m, s, b, r, c);

    const float* predRow = pred + (size_t)m * DD;
    const float* tgtRow  = enc + (size_t)(b * 49 + (s + 1 + r) * 7 + c) * DD;

    int nr = 0;
    if (lane < NNEG)
        nr = neg_idx[((((b * 5 + s) * 6 + r) * 7 + c) * NNEG) + lane];
    int negrow[NNEG];
    #pragma unroll
    for (int i = 0; i < NNEG; i++) negrow[i] = __shfl(nr, i, 32);

    float accv[17];
    #pragma unroll
    for (int i = 0; i < 17; i++) accv[i] = 0.0f;

    for (int it = 0; it < DD / 128; it++) {
        int k = it * 128 + lane * 4;
        float4 pv = *(const float4*)(predRow + k);
        float4 tv = *(const float4*)(tgtRow + k);
        accv[0] += pv.x * tv.x + pv.y * tv.y + pv.z * tv.z + pv.w * tv.w;
        #pragma unroll
        for (int n = 0; n < NNEG; n++) {
            float4 nv = *(const float4*)(enc + (size_t)negrow[n] * DD + k);
            accv[n + 1] += pv.x * nv.x + pv.y * nv.y + pv.z * nv.z + pv.w * nv.w;
        }
    }
    #pragma unroll
    for (int i = 0; i < 17; i++) {
        float v = accv[i];
        #pragma unroll
        for (int off = 16; off > 0; off >>= 1) v += __shfl_xor(v, off, 32);
        accv[i] = v;
    }

    __shared__ float2 wres[8];
    if (lane == 0) {
        float mx = accv[0];
        #pragma unroll
        for (int i = 1; i < 17; i++) mx = fmaxf(mx, accv[i]);
        float sum = 0.0f;
        #pragma unroll
        for (int i = 0; i < 17; i++) sum += expf(accv[i] - mx);
        float loss = logf(sum) + mx - accv[0];
        float corr = (accv[0] >= mx) ? 1.0f : 0.0f;   // ties -> argmax==0
        wres[wid] = make_float2(loss, corr);
    }
    __syncthreads();
    if (tid == 0) {
        float l = 0.0f, cc = 0.0f;
        #pragma unroll
        for (int i = 0; i < 8; i++) { l += wres[i].x; cc += wres[i].y; }
        partial[blockIdx.x * 2]     = l;
        partial[blockIdx.x * 2 + 1] = cc;
    }
}

// ---------------------------------------------------------------------------
// Kernel 4: final deterministic reduction of 1120 block partials
__global__ __launch_bounds__(256)
void reduce_kernel(const float* __restrict__ partial, float* __restrict__ out) {
    __shared__ float sl[256], sc[256];
    float l = 0.0f, c = 0.0f;
    for (int i = threadIdx.x; i < 1120; i += 256) {
        l += partial[2 * i];
        c += partial[2 * i + 1];
    }
    sl[threadIdx.x] = l; sc[threadIdx.x] = c;
    __syncthreads();
    for (int off = 128; off > 0; off >>= 1) {
        if (threadIdx.x < off) {
            sl[threadIdx.x] += sl[threadIdx.x + off];
            sc[threadIdx.x] += sc[threadIdx.x + off];
        }
        __syncthreads();
    }
    if (threadIdx.x == 0) {
        out[0] = sl[0] / (float)NPRED;
        out[1] = sc[0] / (float)NPRED;
    }
}

// ---------------------------------------------------------------------------
extern "C" void kernel_launch(void* const* d_in, const int* in_sizes, int n_in,
                              void* d_out, int out_size, void* d_ws, size_t ws_size,
                              hipStream_t stream) {
    const float* contexts  = (const float*)d_in[0];
    const float* encodings = (const float*)d_in[1];
    const float* Wk_w      = (const float*)d_in[2];
    const float* Wk_b      = (const float*)d_in[3];
    const int*   neg_idx   = (const int*)d_in[4];
    float* out = (float*)d_out;

    // workspace layout (bytes):
    //   Wbf    : 5*1280*1280*2  = 16,384,000
    //   Apack  : 8960*1280*2    = 22,937,600
    //   pred   : 8960*1280*4    = 45,875,200
    //   partial: 1120*2*4       =      8,960   (total ~85.2 MB)
    char* ws = (char*)d_ws;
    unsigned short* Wbf     = (unsigned short*)ws;
    unsigned short* Apack   = (unsigned short*)(ws + 16384000);
    float*          pred    = (float*)(ws + 16384000 + 22937600);
    float*          partial = (float*)(ws + 16384000 + 22937600 + 45875200);

    // (8,192,000 + 11,468,800) elems / 8 per thread / 256 per block = 9600
    cvt_kernel<<<9600, 256, 0, stream>>>(Wk_w, contexts, Wbf, Apack);
    // N tiles = 1280/256 = 5 ; max M tiles = ceil(2688/128) = 21 ; z = step
    gemm_kernel<<<dim3(5, 21, 5), 256, 0, stream>>>(Apack, Wbf, Wk_b, pred);
    score_kernel<<<1120, 256, 0, stream>>>(pred, encodings, neg_idx, partial);
    reduce_kernel<<<1, 256, 0, stream>>>(partial, out);
}
